// Generator_42193758715996
// MI455X (gfx1250) — matrix operand order repaired and checked by reading, not compiled, not run on hardware
//
#include <hip/hip_runtime.h>
#include <math.h>

// ---------- types ----------
typedef __attribute__((ext_vector_type(16))) __bf16 v16bf;
typedef __attribute__((ext_vector_type(8)))  float  v8f;

union BF16x16 { v16bf v; uint4 u[2]; };

__device__ __forceinline__ unsigned short f2bf(float f) {
  union { float f; unsigned u; } a; a.f = f;
  unsigned r = a.u + 0x7fffu + ((a.u >> 16) & 1u);   // round-to-nearest-even
  return (unsigned short)(r >> 16);
}

// Constants
#define BB 128
#define TT 512
#define HH 512
#define K0 1152   // 64 xt + 512 cond + 2 prev + 62 pad + 512 h0
#define K1 1024   // 512 h0 + 512 h1
#define N4H 2048

// ---------- setup kernels ----------
__global__ void build_w0(const float* __restrict__ wih0, const float* __restrict__ whh0,
                         unsigned short* __restrict__ W0) {
  int idx = blockIdx.x * 256 + threadIdx.x;
  if (idx >= N4H * K0) return;
  int n = idx / K0, k = idx - n * K0;
  float v = (k < 578) ? wih0[(size_t)n * 578 + k]
                      : ((k < 640) ? 0.0f : whh0[(size_t)n * 512 + (k - 640)]);
  W0[idx] = f2bf(v);
}

__global__ void build_w1(const float* __restrict__ wih1, const float* __restrict__ whh1,
                         unsigned short* __restrict__ W1) {
  int idx = blockIdx.x * 256 + threadIdx.x;
  if (idx >= N4H * K1) return;
  int n = idx >> 10, k = idx & 1023;
  float v = (k < 512) ? wih1[(size_t)n * 512 + k] : whh1[(size_t)n * 512 + (k - 512)];
  W1[idx] = f2bf(v);
}

__global__ void build_wo1t(const float* __restrict__ wo1, unsigned short* __restrict__ Wt) {
  int idx = blockIdx.x * 256 + threadIdx.x;
  if (idx >= 256 * 512) return;
  int n = idx >> 9, k = idx & 511;
  Wt[idx] = f2bf(wo1[(size_t)k * 256 + n]);
}

__global__ void bias_sum(const float* b0a, const float* b0b, const float* b1a,
                         const float* b1b, float* s0, float* s1) {
  int idx = blockIdx.x * 256 + threadIdx.x;
  if (idx >= N4H) return;
  s0[idx] = b0a[idx] + b0b[idx];
  s1[idx] = b1a[idx] + b1b[idx];
}

__global__ void emb_gather(const int* __restrict__ char_ids, const float* __restrict__ emb,
                           unsigned short* __restrict__ xsbf) {
  int idx = blockIdx.x * 256 + threadIdx.x;
  if (idx >= TT * BB * 64) return;
  int t = idx >> 13;          // / (128*64)
  int r = idx & 8191;
  int b = r >> 6, e = r & 63;
  int cid = char_ids[b * TT + t];
  xsbf[idx] = f2bf(emb[cid * 64 + e]);
}

// Linear -> LayerNorm -> LeakyReLU(0.2). One block per batch row.
__global__ __launch_bounds__(256) void cond_layer(
    const float* __restrict__ x, int K, const float* __restrict__ w,
    const float* __restrict__ bias, const float* __restrict__ gamma,
    const float* __restrict__ beta, float* __restrict__ outf,
    unsigned short* __restrict__ outbf, int obStride, int obOff) {
  __shared__ float vals[512];
  __shared__ float red[256];
  int b = blockIdx.x, tid = threadIdx.x;
  const float* xr = x + (size_t)b * K;
  for (int j = tid; j < 512; j += 256) {
    float s = bias[j];
    for (int k = 0; k < K; ++k) s = fmaf(xr[k], w[(size_t)k * 512 + j], s);
    vals[j] = s;
  }
  __syncthreads();
  float v0 = vals[tid], v1 = vals[tid + 256];
  red[tid] = v0 + v1;
  __syncthreads();
  for (int st = 128; st; st >>= 1) { if (tid < st) red[tid] += red[tid + st]; __syncthreads(); }
  float mean = red[0] * (1.0f / 512.0f);
  __syncthreads();
  float d0 = v0 - mean, d1 = v1 - mean;
  red[tid] = d0 * d0 + d1 * d1;
  __syncthreads();
  for (int st = 128; st; st >>= 1) { if (tid < st) red[tid] += red[tid + st]; __syncthreads(); }
  float rstd = rsqrtf(red[0] * (1.0f / 512.0f) + 1e-5f);
  float y0 = d0 * rstd * gamma[tid] + beta[tid];
  float y1 = d1 * rstd * gamma[tid + 256] + beta[tid + 256];
  y0 = (y0 > 0.f) ? y0 : 0.2f * y0;
  y1 = (y1 > 0.f) ? y1 : 0.2f * y1;
  if (outf) {
    outf[(size_t)b * 512 + tid] = y0;
    outf[(size_t)b * 512 + tid + 256] = y1;
  }
  if (outbf) {
    outbf[(size_t)b * obStride + obOff + tid] = f2bf(y0);
    outbf[(size_t)b * obStride + obOff + tid + 256] = f2bf(y1);
  }
}

// Zero c0/c1, X1, and fill X0 xt(t=0)/prev/pad/h0 regions (cond region untouched).
__global__ void init_state(float* c0, float* c1, unsigned short* X0, unsigned short* X1,
                           const unsigned short* __restrict__ xsbf,
                           const float* __restrict__ init_timing) {
  int tid0 = blockIdx.x * 256 + threadIdx.x;
  int nthr = gridDim.x * 256;
  for (int i = tid0; i < BB * HH; i += nthr) { c0[i] = 0.0f; c1[i] = 0.0f; }
  for (int i = tid0; i < BB * K1; i += nthr) X1[i] = 0;
  for (int i = tid0; i < BB * K0; i += nthr) {
    int b = i / K0, k = i - b * K0;
    unsigned short v;
    if (k < 64)       v = xsbf[b * 64 + k];          // xt for t=0
    else if (k < 576) continue;                      // cond (written by cond_layer)
    else if (k == 576) v = f2bf(init_timing[0]);
    else if (k == 577) v = f2bf(init_timing[1]);
    else               v = 0;                        // pad + h0
    X0[i] = v;
  }
}

// ---------- WMMA tile helpers (contiguous k order, same for A and B) ----------
__device__ __forceinline__ v16bf load_a16x32(const unsigned short* rowbase, int sel, int kc) {
  // A 16x32 bf16: lanes0-15 -> K 0..15; lanes16-31 -> K 16..31 (contiguous per row)
  BF16x16 t;
  const unsigned short* p = rowbase + kc + sel * 16;
  t.u[0] = *(const uint4*)p;
  t.u[1] = *(const uint4*)(p + 8);
  return t.v;
}
__device__ __forceinline__ v16bf load_b32x16(const unsigned short* wrow, int sel, int kc) {
  // B 32x16 bf16: lanes0-15 -> K 0..15; lanes16-31 -> K 16..31 (contiguous per row)
  BF16x16 t;
  const unsigned short* p = wrow + kc + sel * 16;
  t.u[0] = *(const uint4*)p;
  t.u[1] = *(const uint4*)(p + 8);
  return t.v;
}

// ---------- gate GEMM: out(128 x N) = X(128 x K) * W(N x K)^T ----------
// Register-blocked: wave = 1 mTile x 4 nTiles (A fragment reused 4x, 4 accumulators).
// Block = 8 waves = all 8 mTiles x same 4 nTiles; grid = N/64 blocks, so every W
// element is read from L2 exactly once per step. No LDS, no barriers in hot loop.
__global__ __launch_bounds__(256) void gates_gemm(
    const unsigned short* __restrict__ X, const unsigned short* __restrict__ W,
    float* __restrict__ out, int K, int N) {
  int tid = threadIdx.x;
  int wave = tid >> 5, lane = tid & 31;
  int rsel = lane & 15, sel = lane >> 4;
  int mTile = wave;                 // 8 waves cover M = 128
  int nbase = blockIdx.x * 4;       // 4 nTiles per wave
  const unsigned short* arow = X + (size_t)(mTile * 16 + rsel) * K;
  const unsigned short* wb   = W + (size_t)(nbase * 16 + rsel) * K;
  const size_t wstride = (size_t)16 * K;   // one nTile of W rows

  v8f z = {0.f, 0.f, 0.f, 0.f, 0.f, 0.f, 0.f, 0.f};
  v8f acc0 = z, acc1 = z, acc2 = z, acc3 = z;

  for (int kc = 0; kc < K; kc += 32) {
    v16bf a = load_a16x32(arow, sel, kc);
    v16bf b0 = load_b32x16(wb,               sel, kc);
    v16bf b1 = load_b32x16(wb + wstride,     sel, kc);
    v16bf b2 = load_b32x16(wb + 2 * wstride, sel, kc);
    v16bf b3 = load_b32x16(wb + 3 * wstride, sel, kc);
    acc0 = __builtin_amdgcn_wmma_f32_16x16x32_bf16(false, a, false, b0, (short)0, acc0, false, false);
    acc1 = __builtin_amdgcn_wmma_f32_16x16x32_bf16(false, a, false, b1, (short)0, acc1, false, false);
    acc2 = __builtin_amdgcn_wmma_f32_16x16x32_bf16(false, a, false, b2, (short)0, acc2, false, false);
    acc3 = __builtin_amdgcn_wmma_f32_16x16x32_bf16(false, a, false, b3, (short)0, acc3, false, false);
  }

  // C/D layout: VGPR i -> M = sel*8 + i, N = lane&15
  int mrow = mTile * 16 + sel * 8;
  v8f accs[4] = {acc0, acc1, acc2, acc3};
#pragma unroll
  for (int j = 0; j < 4; ++j) {
    int n = (nbase + j) * 16 + rsel;
    size_t base = (size_t)mrow * N + n;
#pragma unroll
    for (int i = 0; i < 8; ++i) out[base + (size_t)i * N] = accs[j][i];
  }
}

// ---------- LSTM pointwise: gates -> c,h ; h written as bf16 into A buffers ----------
__global__ __launch_bounds__(256) void lstm_pointwise(
    const float* __restrict__ gates, const float* __restrict__ bsum, float* __restrict__ c,
    unsigned short* __restrict__ dA, int ldA, int offA,
    unsigned short* __restrict__ dB, int ldB, int offB) {
  int idx = blockIdx.x * 256 + threadIdx.x;
  if (idx >= BB * HH) return;
  int b = idx >> 9, h = idx & 511;
  const float* g = gates + (size_t)b * N4H;
  float gi = g[h] + bsum[h];
  float gf = g[512 + h] + bsum[512 + h];
  float gg = g[1024 + h] + bsum[1024 + h];
  float go = g[1536 + h] + bsum[1536 + h];
  float si = 1.0f / (1.0f + expf(-gi));
  float sf = 1.0f / (1.0f + expf(-gf));
  float so = 1.0f / (1.0f + expf(-go));
  float cn = sf * c[idx] + si * tanhf(gg);
  c[idx] = cn;
  unsigned short hb = f2bf(so * tanhf(cn));
  dA[(size_t)b * ldA + offA + h] = hb;
  if (dB) dB[(size_t)b * ldB + offB + h] = hb;
}

// ---------- fused output head + next-step feed ----------
__global__ __launch_bounds__(256) void out_head(
    const unsigned short* __restrict__ X1, const unsigned short* __restrict__ Wo1T,
    const float* __restrict__ bo1, const float* __restrict__ wo2,
    const float* __restrict__ bo2, const unsigned short* __restrict__ xsbf,
    unsigned short* __restrict__ X0, float* __restrict__ out, int t) {
  __shared__ __align__(16) unsigned short Ah[16 * 512];
  __shared__ float tmem[16 * 256];
  int mTile = blockIdx.x, tid = threadIdx.x;
  // write xt for step t+1 into X0 (harmless repeat at t=511)
  int tn = (t + 1 < TT) ? (t + 1) : (TT - 1);
  for (int i = tid; i < 16 * 64; i += 256) {
    int r = i >> 6, e = i & 63;
    int b = mTile * 16 + r;
    X0[(size_t)b * K0 + e] = xsbf[((size_t)tn * BB + b) * 64 + e];
  }
  // stage h1 (X1 cols 512..1023) into LDS
  for (int c = tid; c < 1024; c += 256) {
    int r = c >> 6, g = c & 63;
    *(uint4*)(Ah + r * 512 + g * 8) =
        *(const uint4*)(X1 + (size_t)(mTile * 16 + r) * K1 + 512 + g * 8);
  }
  __syncthreads();
  int wave = tid >> 5, lane = tid & 31;
  int rsel = lane & 15, sel = lane >> 4;
  int nt0 = wave * 2, nt1 = nt0 + 1;
  const unsigned short* w0r = Wo1T + (size_t)(nt0 * 16 + rsel) * 512;
  const unsigned short* w1r = Wo1T + (size_t)(nt1 * 16 + rsel) * 512;
  const unsigned short* ar = Ah + rsel * 512;
  v8f acc0 = {0.f, 0.f, 0.f, 0.f, 0.f, 0.f, 0.f, 0.f};
  v8f acc1 = {0.f, 0.f, 0.f, 0.f, 0.f, 0.f, 0.f, 0.f};
  for (int kc = 0; kc < 512; kc += 32) {
    v16bf a = load_a16x32(ar, sel, kc);
    v16bf b0 = load_b32x16(w0r, sel, kc);
    v16bf b1 = load_b32x16(w1r, sel, kc);
    acc0 = __builtin_amdgcn_wmma_f32_16x16x32_bf16(false, a, false, b0, (short)0, acc0,
                                                   false, false);
    acc1 = __builtin_amdgcn_wmma_f32_16x16x32_bf16(false, a, false, b1, (short)0, acc1,
                                                   false, false);
  }
  int n0 = nt0 * 16 + rsel, n1 = nt1 * 16 + rsel;
  float bb0 = bo1[n0], bb1 = bo1[n1];
#pragma unroll
  for (int i = 0; i < 8; ++i) {
    int m = sel * 8 + i;
    float xa = acc0[i] + bb0; xa = (xa > 0.f) ? xa : 0.2f * xa;
    float xb = acc1[i] + bb1; xb = (xb > 0.f) ? xb : 0.2f * xb;
    tmem[m * 256 + n0] = xa;
    tmem[m * 256 + n1] = xb;
  }
  __syncthreads();
  if (tid < 32) {
    int r = tid & 15, col = tid >> 4;
    float s = bo2[col];
    for (int k = 0; k < 256; ++k) s = fmaf(tmem[r * 256 + k], wo2[k * 2 + col], s);
    float sp = fmaxf(s, 0.f) + log1pf(expf(-fabsf(s))) + 0.005f;  // softplus + 0.005
    int b = mTile * 16 + r;
    out[(size_t)b * (TT * 2) + (size_t)t * 2 + col] = sp;
    X0[(size_t)b * K0 + 576 + col] = f2bf(sp);                    // prev feedback
  }
}

// ---------- host ----------
extern "C" void kernel_launch(void* const* d_in, const int* in_sizes, int n_in,
                              void* d_out, int out_size, void* d_ws, size_t ws_size,
                              hipStream_t stream) {
  const int*   char_ids = (const int*)d_in[0];
  const float* z      = (const float*)d_in[1];
  const float* emb    = (const float*)d_in[2];
  const float* cw1    = (const float*)d_in[3];
  const float* cb1    = (const float*)d_in[4];
  const float* cg1    = (const float*)d_in[5];
  const float* cbeta1 = (const float*)d_in[6];
  const float* cw2    = (const float*)d_in[7];
  const float* cb2    = (const float*)d_in[8];
  const float* cg2    = (const float*)d_in[9];
  const float* cbeta2 = (const float*)d_in[10];
  const float* wih0   = (const float*)d_in[11];
  const float* whh0   = (const float*)d_in[12];
  const float* bih0   = (const float*)d_in[13];
  const float* bhh0   = (const float*)d_in[14];
  const float* wih1   = (const float*)d_in[15];
  const float* whh1   = (const float*)d_in[16];
  const float* bih1   = (const float*)d_in[17];
  const float* bhh1   = (const float*)d_in[18];
  const float* wo1    = (const float*)d_in[19];
  const float* bo1    = (const float*)d_in[20];
  const float* wo2    = (const float*)d_in[21];
  const float* bo2    = (const float*)d_in[22];
  const float* init_timing = (const float*)d_in[23];

  char* ws = (char*)d_ws;
  size_t off = 0;
  auto alloc = [&](size_t bytes) -> void* {
    void* p = ws + off;
    off = (off + bytes + 255) & ~(size_t)255;
    return p;
  };
  unsigned short* W0    = (unsigned short*)alloc((size_t)N4H * K0 * 2);
  unsigned short* W1    = (unsigned short*)alloc((size_t)N4H * K1 * 2);
  unsigned short* Wo1T  = (unsigned short*)alloc((size_t)256 * 512 * 2);
  unsigned short* xsbf  = (unsigned short*)alloc((size_t)TT * BB * 64 * 2);
  unsigned short* X0    = (unsigned short*)alloc((size_t)BB * K0 * 2);
  unsigned short* X1    = (unsigned short*)alloc((size_t)BB * K1 * 2);
  float* cond1  = (float*)alloc((size_t)BB * 512 * 4);
  float* gates0 = (float*)alloc((size_t)BB * N4H * 4);
  float* gates1 = (float*)alloc((size_t)BB * N4H * 4);
  float* c0     = (float*)alloc((size_t)BB * HH * 4);
  float* c1     = (float*)alloc((size_t)BB * HH * 4);
  float* b0sum  = (float*)alloc((size_t)N4H * 4);
  float* b1sum  = (float*)alloc((size_t)N4H * 4);
  float* out = (float*)d_out;

  // one-time setup (re-run every call; deterministic)
  build_w0<<<(N4H * K0 + 255) / 256, 256, 0, stream>>>(wih0, whh0, W0);
  build_w1<<<(N4H * K1 + 255) / 256, 256, 0, stream>>>(wih1, whh1, W1);
  build_wo1t<<<(256 * 512 + 255) / 256, 256, 0, stream>>>(wo1, Wo1T);
  bias_sum<<<(N4H + 255) / 256, 256, 0, stream>>>(bih0, bhh0, bih1, bhh1, b0sum, b1sum);
  emb_gather<<<(TT * BB * 64 + 255) / 256, 256, 0, stream>>>(char_ids, emb, xsbf);
  cond_layer<<<BB, 256, 0, stream>>>(z, 128, cw1, cb1, cg1, cbeta1, cond1,
                                     (unsigned short*)nullptr, 0, 0);
  cond_layer<<<BB, 256, 0, stream>>>(cond1, 512, cw2, cb2, cg2, cbeta2,
                                     (float*)nullptr, X0, K0, 64);
  init_state<<<256, 256, 0, stream>>>(c0, c1, X0, X1, xsbf, init_timing);

  // sequential scan over T
  for (int t = 0; t < TT; ++t) {
    gates_gemm<<<dim3(N4H / 64), 256, 0, stream>>>(X0, W0, gates0, K0, N4H);
    lstm_pointwise<<<(BB * HH) / 256, 256, 0, stream>>>(gates0, b0sum, c0,
                                                        X0, K0, 640, X1, K1, 0);
    gates_gemm<<<dim3(N4H / 64), 256, 0, stream>>>(X1, W1, gates1, K1, N4H);
    lstm_pointwise<<<(BB * HH) / 256, 256, 0, stream>>>(gates1, b1sum, c1,
                                                        X1, K1, 512,
                                                        (unsigned short*)nullptr, 0, 0);
    out_head<<<8, 256, 0, stream>>>(X1, Wo1T, bo1, wo2, bo2, xsbf, X0, out, t);
  }
}
